// CrossDocumentAttention_4166118277821
// MI455X (gfx1250) — matrix operand order repaired and checked by reading, compiled, not run on hardware
//
#include <hip/hip_runtime.h>
#include <hip/hip_bf16.h>

typedef __attribute__((ext_vector_type(16))) _Float16 v16h;
typedef __attribute__((ext_vector_type(8)))  _Float16 v8h;
typedef __attribute__((ext_vector_type(4)))  _Float16 v4h;
typedef __attribute__((ext_vector_type(8)))  float    v8f;
typedef __attribute__((ext_vector_type(4)))  unsigned int u32x4;
typedef __attribute__((ext_vector_type(8)))  int      i32x8;
typedef __attribute__((ext_vector_type(4)))  int      i32x4;

#define D_DIM   1024
#define BM      128      // query rows per block (8 waves x 16)
#define BN      64       // keys per tile
#define BD      256      // output columns per block
#define SCALE   0.03125f // 1/sqrt(1024)
#define VT_STR  72       // padded LDS stride (144B, 16B-aligned)

// LDS partition (halves): VR (TDM row-major V tile) | VT (transposed) | P
#define VR_HALVES (BN * BD)          // 16384
#define VT_HALVES (BD * VT_STR)      // 18432
#define P_HALVES  (8 * 16 * 64)      // 8192
#define SMEM_BYTES ((VR_HALVES + VT_HALVES + P_HALVES) * 2)  // 86016 B

#if defined(__has_builtin)
#if __has_builtin(__builtin_amdgcn_tensor_load_to_lds) && \
    __has_builtin(__builtin_amdgcn_s_wait_tensorcnt)
#define HAVE_TDM 1
#endif
#endif

// ---------------- Pass 1: f32 -> f16 convert ----------------
__global__ void xda_cvt_f16(const float* __restrict__ x,
                            _Float16* __restrict__ xh, int n4) {
    int i = blockIdx.x * blockDim.x + threadIdx.x;
    if (i < n4) {
        float4 f = ((const float4*)x)[i];
        v4h h = { (_Float16)f.x, (_Float16)f.y, (_Float16)f.z, (_Float16)f.w };
        ((v4h*)xh)[i] = h;
    }
}

#ifdef HAVE_TDM
// Issue one TDM DMA: 64x256 f16 tile (rows = keys, stride D_DIM) -> LDS offset 0.
__device__ __forceinline__ void tdm_load_vtile(const _Float16* gsrc, int N) {
    unsigned long long ga = (unsigned long long)(uintptr_t)gsrc;
    // D# group 0: count=1 | lds_addr(=0) | global_addr | type=2
    u32x4 g0 = { 1u,
                 0u,
                 (unsigned)(ga & 0xFFFFFFFFull),
                 (unsigned)((ga >> 32) & 0x01FFFFFFull) | (2u << 30) };
    // D# group 1: data_size=2B; tensor_dim0=D_DIM; tensor_dim1=N;
    // tile_dim0=BD(cols); tile_dim1=BN(keys); tensor_dim0_stride=D_DIM
    i32x8 g1 = { (int)(1u << 16),
                 (int)(((unsigned)D_DIM & 0xFFFFu) << 16),
                 (int)((((unsigned)D_DIM >> 16) & 0xFFFFu) |
                       (((unsigned)N & 0xFFFFu) << 16)),
                 (int)((((unsigned)N >> 16) & 0xFFFFu) | ((unsigned)BD << 16)),
                 (int)BN,
                 (int)D_DIM,
                 0, 0 };
    i32x4 z4 = { 0, 0, 0, 0 };
#if __clang_major__ >= 23
    i32x8 z8 = { 0, 0, 0, 0, 0, 0, 0, 0 };
    __builtin_amdgcn_tensor_load_to_lds(g0, g1, z4, z4, z8, 0);
#else
    __builtin_amdgcn_tensor_load_to_lds(g0, g1, z4, z4, 0);
#endif
}
#endif

// ---------------- Pass 2: flash attention with WMMA ----------------
__global__ __launch_bounds__(256) void xda_attn(const _Float16* __restrict__ xh,
                                                const int* __restrict__ doc,
                                                float* __restrict__ out, int N) {
    extern __shared__ __align__(16) _Float16 smem[];
    _Float16* VR = smem;                 // TDM landing: [key][col] row-major
    _Float16* VT = smem + VR_HALVES;     // transposed: [col][key], padded stride
    _Float16* P  = VT + VT_HALVES;       // per-wave probability tiles

    const int lane = threadIdx.x & 31;
    const int wave = threadIdx.x >> 5;
    const int g    = lane >> 4;
    const int ln   = lane & 15;
    const int r0   = blockIdx.x * BM + wave * 16;
    const int c0   = blockIdx.y * BD;

    int rowdoc[8];
#pragma unroll
    for (int r = 0; r < 8; ++r) rowdoc[r] = doc[r0 + g * 8 + r];

    float mrow[8], lrow[8];
#pragma unroll
    for (int r = 0; r < 8; ++r) { mrow[r] = -1e30f; lrow[r] = 0.0f; }

    v8f acc[16];
#pragma unroll
    for (int o = 0; o < 16; ++o) acc[o] = (v8f)0.0f;

    const _Float16* qbase = xh + (size_t)(r0 + ln) * D_DIM + (g ? 8 : 0);
    _Float16* Pw = P + wave * (16 * 64);

    for (int k0 = 0; k0 < N; k0 += BN) {
        __syncthreads();   // previous tile's VT/VR reads complete

        // prefetch next key tile's K rows into L2/WGP$
        if (k0 + BN < N)
            __builtin_prefetch(xh + (size_t)(k0 + BN) * D_DIM + threadIdx.x * 256, 0, 1);

#ifdef HAVE_TDM
        // ---- stage V tile via Tensor Data Mover, then transpose in LDS ----
        if (wave == 0) {
            tdm_load_vtile(xh + (size_t)k0 * D_DIM + c0, N);
            __builtin_amdgcn_s_wait_tensorcnt(0);
        }
        __syncthreads();   // VR ready for everyone
        {
            const int t = threadIdx.x;
#pragma unroll
            for (int j = 0; j < 8; ++j) {
                int c8  = t + 256 * j;
                int key = c8 >> 5;
                int cb  = (c8 & 31) * 8;
                v8h v = *(const v8h*)(VR + key * BD + cb);
#pragma unroll
                for (int e = 0; e < 8; ++e) VT[(cb + e) * VT_STR + key] = v[e];
            }
        }
#else
        // ---- fallback: stage V tile transposed straight from global ----
        {
            const int t = threadIdx.x;
#pragma unroll
            for (int j = 0; j < 8; ++j) {
                int c8  = t + 256 * j;
                int key = c8 >> 5;
                int cb  = (c8 & 31) * 8;
                v8h v = *(const v8h*)(xh + (size_t)(k0 + key) * D_DIM + c0 + cb);
#pragma unroll
                for (int e = 0; e < 8; ++e) VT[(cb + e) * VT_STR + key] = v[e];
            }
        }
#endif
        __syncthreads();

        // ---- S = Q K^T for 16x64 tile, f32 accumulate over D ----
        v8f s[4];
#pragma unroll
        for (int sub = 0; sub < 4; ++sub) s[sub] = (v8f)0.0f;

#pragma unroll 4
        for (int d0 = 0; d0 < D_DIM; d0 += 32) {
            v16h a;
            ((v8h*)&a)[0] = *(const v8h*)(qbase + d0);
            ((v8h*)&a)[1] = *(const v8h*)(qbase + d0 + 16);
#pragma unroll
            for (int sub = 0; sub < 4; ++sub) {
                const _Float16* kp =
                    xh + (size_t)(k0 + sub * 16 + ln) * D_DIM + d0 + g * 8;
                v16h b;
                ((v8h*)&b)[0] = *(const v8h*)(kp);
                ((v8h*)&b)[1] = *(const v8h*)(kp + 16);
                s[sub] = __builtin_amdgcn_wmma_f32_16x16x32_f16(
                    false, a, false, b, (short)0, s[sub], false, false);
            }
        }

        // ---- additive doc bias + online softmax update ----
        int coldoc[4];
#pragma unroll
        for (int sub = 0; sub < 4; ++sub) coldoc[sub] = doc[k0 + sub * 16 + ln];

#pragma unroll
        for (int r = 0; r < 8; ++r) {
            float v0 = s[0][r] * SCALE + ((rowdoc[r] == coldoc[0]) ? 1.0f : 0.0f);
            float v1 = s[1][r] * SCALE + ((rowdoc[r] == coldoc[1]) ? 1.0f : 0.0f);
            float v2 = s[2][r] * SCALE + ((rowdoc[r] == coldoc[2]) ? 1.0f : 0.0f);
            float v3 = s[3][r] * SCALE + ((rowdoc[r] == coldoc[3]) ? 1.0f : 0.0f);
            float tm = fmaxf(fmaxf(v0, v1), fmaxf(v2, v3));
#pragma unroll
            for (int m = 1; m < 16; m <<= 1)
                tm = fmaxf(tm, __shfl_xor(tm, m, 32));
            float mn   = fmaxf(mrow[r], tm);
            float corr = __expf(mrow[r] - mn);
            mrow[r] = mn;
            float p0 = __expf(v0 - mn), p1 = __expf(v1 - mn);
            float p2 = __expf(v2 - mn), p3 = __expf(v3 - mn);
            float ts = (p0 + p1) + (p2 + p3);
#pragma unroll
            for (int m = 1; m < 16; m <<= 1)
                ts += __shfl_xor(ts, m, 32);
            lrow[r] = lrow[r] * corr + ts;
#pragma unroll
            for (int o = 0; o < 16; ++o) acc[o][r] *= corr;
            int mrowi = (r + 8 * g) * 64;
            Pw[mrowi +  0 + ln] = (_Float16)p0;
            Pw[mrowi + 16 + ln] = (_Float16)p1;
            Pw[mrowi + 32 + ln] = (_Float16)p2;
            Pw[mrowi + 48 + ln] = (_Float16)p3;
        }

        // ---- acc += P(16x64) @ V(64x256) ----
#pragma unroll
        for (int kt = 0; kt < 2; ++kt) {
            v16h a;
            const _Float16* pp = Pw + ln * 64 + kt * 32 + (g ? 8 : 0);
            ((v8h*)&a)[0] = *(const v8h*)(pp);
            ((v8h*)&a)[1] = *(const v8h*)(pp + 16);
#pragma unroll
            for (int o = 0; o < 16; ++o) {
                const _Float16* vp = VT + (o * 16 + ln) * VT_STR + kt * 32 + g * 8;
                v16h b;
                ((v8h*)&b)[0] = *(const v8h*)(vp);
                ((v8h*)&b)[1] = *(const v8h*)(vp + 16);
                acc[o] = __builtin_amdgcn_wmma_f32_16x16x32_f16(
                    false, a, false, b, (short)0, acc[o], false, false);
            }
        }
    }

    // ---- epilogue: normalize and store f32 ----
#pragma unroll
    for (int r = 0; r < 8; ++r) {
        float inv = 1.0f / lrow[r];
        int row = r0 + g * 8 + r;
#pragma unroll
        for (int o = 0; o < 16; ++o)
            out[(size_t)row * D_DIM + c0 + o * 16 + ln] = acc[o][r] * inv;
    }
}

extern "C" void kernel_launch(void* const* d_in, const int* in_sizes, int n_in,
                              void* d_out, int out_size, void* d_ws, size_t ws_size,
                              hipStream_t stream) {
    const float* x   = (const float*)d_in[0];
    const int*   doc = (const int*)d_in[1];
    float*       out = (float*)d_out;
    _Float16*    xh  = (_Float16*)d_ws;     // needs N*D*2 = 16 MB

    const int N  = in_sizes[1];             // 8192
    const int n4 = (N * D_DIM) / 4;

    xda_cvt_f16<<<(n4 + 255) / 256, 256, 0, stream>>>(x, xh, n4);

    dim3 grid(N / BM, D_DIM / BD);
    xda_attn<<<grid, 256, SMEM_BYTES, stream>>>(xh, doc, out, N);
}
